// ElmobiLm_5669356834277
// MI455X (gfx1250) — matrix operand order repaired and checked by reading, not compiled, not run on hardware
//
#include <hip/hip_runtime.h>
#include <hip/hip_bf16.h>

// Problem constants (from reference setup_inputs):
#define B_   16
#define T_   128
#define D_   512
#define H_   512
#define C_   4096
#define G4C  (4 * C_)      // 16384 gate columns
#define L_   2
#define NWG  64            // persistent workgroups in recurrent kernel
#define OUT_LD 1024        // d_out innermost stride (2*H)

typedef __attribute__((ext_vector_type(16))) __bf16 v16bf;
typedef __attribute__((ext_vector_type(8)))  __bf16 v8bf;
typedef __attribute__((ext_vector_type(8)))  float  v8f;
typedef __attribute__((ext_vector_type(4)))  unsigned int v4u;
typedef __attribute__((ext_vector_type(8)))  int v8i;
typedef __attribute__((ext_vector_type(4)))  int v4i;

typedef __attribute__((address_space(3))) const char lds_cchar;

// ---------------------------------------------------------------- WMMA helpers
__device__ __forceinline__ v8f wmma_bf16(v16bf a, v16bf b, v8f c) {
  // D = A(16x32 bf16) * B(32x16 bf16) + C(16x16 f32)
  return __builtin_amdgcn_wmma_f32_16x16x32_bf16(false, a, false, b,
                                                 (short)0, c, false, false);
}

// A-fragment (16x32 bf16). Lane L: row m = L&15.
// Lanes 0-15 hold K = kb+[0..15]; lanes 16-31 K = kb+[16..31].
__device__ __forceinline__ v16bf load_a_bf16(const __bf16* __restrict__ base, int ld,
                                             int m, int kb, bool hi) {
  const __bf16* p = base + (long)m * ld + kb + (hi ? 16 : 0);
  v8bf lo = *(const v8bf*)p;
  v8bf hh = *(const v8bf*)(p + 8);
  v16bf r;
#pragma unroll
  for (int e = 0; e < 8; ++e) { r[e] = lo[e]; r[8 + e] = hh[e]; }
  return r;
}

// Same A layout but sourced from fp32 (on-the-fly bf16 conversion).
__device__ __forceinline__ v16bf load_a_f32(const float* __restrict__ base, long ld,
                                            int m, int kb, bool hi) {
  const float* p = base + (long)m * ld + kb + (hi ? 16 : 0);
  v16bf r;
#pragma unroll
  for (int e = 0; e < 8; ++e) { r[e] = (__bf16)p[e]; r[8 + e] = (__bf16)p[8 + e]; }
  return r;
}

// B-fragment (32x16 bf16) for B[k,n] = W[nrow, kb+k] with W row-major (N, K):
// lane L holds column n = L&15, 16 contiguous K values starting at kb (+16 for hi lanes).
__device__ __forceinline__ v16bf load_b_bf16(const __bf16* __restrict__ base, int ld,
                                             int nrow, int kb, bool hi) {
  return *(const v16bf*)(base + (long)nrow * ld + kb + (hi ? 16 : 0));
}

// B-fragment from an LDS-staged 128x32 bf16 panel with 80-byte row pitch
// (TDM pads 4 DWORDs after every 16 DWORDs -> conflict-free 16B reads).
__device__ __forceinline__ v16bf load_b_lds(const char* __restrict__ base,
                                            int tile_i, int nl, bool hi) {
  const char* p = base + (tile_i * 16 + nl) * 80 + (hi ? 32 : 0);
  v8bf lo = *(const v8bf*)p;
  v8bf hh = *(const v8bf*)(p + 16);
  v16bf r;
#pragma unroll
  for (int e = 0; e < 8; ++e) { r[e] = lo[e]; r[8 + e] = hh[e]; }
  return r;
}

// ---------------------------------------------------------------- TDM staging
__device__ __forceinline__ unsigned lds_offset_of(const void* p) {
  return (unsigned)(uintptr_t)(lds_cchar*)(const char*)p;
}

// Async-DMA a (nrows x 32) bf16 tile (row pitch 512 elements in memory) into LDS,
// padding the LDS destination to an 80-byte row pitch.  D# per CDNA5 ISA ch.8.
__device__ __forceinline__ void tdm_load_tile(unsigned lds_off,
                                              const __bf16* __restrict__ gsrc,
                                              int nrows) {
  const unsigned long long ga = (unsigned long long)(uintptr_t)gsrc;
  v4u g0;
  g0[0] = 1u;                                        // count=1, user descriptor
  g0[1] = lds_off;                                   // lds_addr
  g0[2] = (unsigned)(ga & 0xffffffffu);              // global_addr[31:0]
  g0[3] = (unsigned)((ga >> 32) & 0x01ffffffu)       // global_addr[56:32]
        | (2u << 30);                                // type=2 ("image")
  v8i g1;
  g1[0] = (1 << 16)                                  // data_size = 2 bytes
        | (1 << 20)                                  // pad_enable
        | (3 << 22)                                  // pad_interval: 16 DWORDs
        | (3 << 25);                                 // pad_amount: 4 DWORDs
  g1[1] = (32 & 0xffff) << 16;                       // tensor_dim0 = 32 (lo16)
  g1[2] = (32 >> 16) | ((nrows & 0xffff) << 16);     // td0 hi | tensor_dim1 lo
  g1[3] = (nrows >> 16) | (32 << 16);                // td1 hi | tile_dim0 = 32
  g1[4] = nrows & 0xffff;                            // tile_dim1 = nrows, tile_dim2 = 0
  g1[5] = 512;                                       // tensor_dim0_stride = 512 elems
  g1[6] = 0;                                         // stride hi | tensor_dim1_stride lo
  g1[7] = 0;
  const v4i z4 = {0, 0, 0, 0};                       // groups 2/3 unused (2-D tile)
  const v8i z8 = {0, 0, 0, 0, 0, 0, 0, 0};           // extra group (unused)
  // 6-arg toolchain form: (g0, g1, g2, g3, gx, cpol)
  __builtin_amdgcn_tensor_load_to_lds(g0, g1, z4, z4, z8, 0);
}

// ---------------------------------------------------------------- grid barrier
__device__ __forceinline__ void grid_sync(unsigned* bar) {
  __syncthreads();
  if (threadIdx.x == 0) {
    __threadfence();
    unsigned old = atomicAdd(bar, 1u);
    unsigned target = (old / NWG + 1u) * NWG;
    while (__hip_atomic_load(bar, __ATOMIC_RELAXED, __HIP_MEMORY_SCOPE_AGENT) < target)
      __builtin_amdgcn_s_sleep(2);
    __threadfence();
  }
  __syncthreads();
}

// ---------------------------------------------------------------- small kernels
__global__ __launch_bounds__(256) void f32_to_bf16_kernel(const float* __restrict__ s,
                                                          __bf16* __restrict__ d, long n) {
  long i = (long)blockIdx.x * blockDim.x + threadIdx.x;
  long stride = (long)gridDim.x * blockDim.x;
  for (; i < n; i += stride) d[i] = (__bf16)s[i];
}

__global__ __launch_bounds__(256) void init_state_kernel(float* h, __bf16* hb,
                                                         float* c, __bf16* a,
                                                         unsigned* bar) {
  int i = blockIdx.x * 256 + threadIdx.x;            // grid: 256 blocks = 65536 threads
  if (i < B_ * C_) { c[i] = 0.f; a[i] = (__bf16)0.f; }
  if (i < B_ * H_) { h[i] = 0.f; hb[i] = (__bf16)0.f; }
  if (i == 0) bar[0] = 0u;
}

// ---------------------------------------------------------------- Xp = in @ Wi.T + b
// M = B*T = 2048, N = 4C = 16384, K = 512.  Block = 128x128 tile, 8 waves,
// wave -> 16 rows x 128 cols.  The 128x32 B panel (shared by all 8 waves) is
// staged in LDS by the Tensor Data Mover, double-buffered: wave 0 issues the
// DMA for k-step i+1 while everyone multiplies k-step i out of LDS.
// All 8 B-fragment LDS loads are issued before the 8 WMMAs so only the first
// matrix op pays a DS wait.
__global__ __launch_bounds__(256) void xproj_gemm_kernel(
    const float* __restrict__ in, long in_ld,
    const __bf16* __restrict__ Wib, const float* __restrict__ bias,
    float* __restrict__ Xp) {
  __shared__ __align__(16) char bstage[2][128 * 80]; // 2 x 10 KB padded panels
  const int lane = threadIdx.x & 31;
  const int wave = threadIdx.x >> 5;
  const bool hi = lane >= 16;
  const int nl = lane & 15;
  const int m0 = blockIdx.y * 128 + wave * 16;       // 16 rows, never crosses batch b
  const int n0 = blockIdx.x * 128;
  const float* Abase = in + (long)m0 * in_ld;
  const __bf16* Bglob = Wib + (long)n0 * D_;         // 128 weight rows at n0

  v8f acc[8];
#pragma unroll
  for (int i = 0; i < 8; ++i) {
    float bv = bias[n0 + i * 16 + nl];
#pragma unroll
    for (int v = 0; v < 8; ++v) acc[i][v] = bv;
  }

  if (wave == 0)
    tdm_load_tile(lds_offset_of(&bstage[0][0]), Bglob, 128);

  for (int kb = 0, it = 0; kb < D_; kb += 32, ++it) {
    const int cur = it & 1;
    if (wave == 0) __builtin_amdgcn_s_wait_tensorcnt(0);
    __syncthreads();                                 // staged panel visible to all
    if (wave == 0 && kb + 32 < D_)
      tdm_load_tile(lds_offset_of(&bstage[cur ^ 1][0]), Bglob + (kb + 32), 128);
    // issue the whole B-fragment batch, then A, then the WMMA chain
    v16bf bfr[8];
#pragma unroll
    for (int i = 0; i < 8; ++i)
      bfr[i] = load_b_lds(&bstage[cur][0], i, nl, hi);
    v16bf a = load_a_f32(Abase, in_ld, nl, kb, hi);
#pragma unroll
    for (int i = 0; i < 8; ++i)
      acc[i] = wmma_bf16(a, bfr[i], acc[i]);
    // no trailing barrier needed: the next write into bstage[cur] is only
    // issued after the *next* iteration's __syncthreads()
  }

  // Scatter to Xp[t][b][n] (original-time layout).
#pragma unroll
  for (int i = 0; i < 8; ++i) {
    const int n = n0 + i * 16 + nl;
#pragma unroll
    for (int v = 0; v < 8; ++v) {
      const int r = m0 + v + (hi ? 8 : 0);           // r = b*T + t
      const int b = r >> 7, t = r & (T_ - 1);
      Xp[((long)t * B_ + b) * G4C + n] = acc[i][v];
    }
  }
}

// ---------------------------------------------------------------- recurrent kernel
// Persistent NWG=64 workgroups.  WG w owns cell chunk j in [w*64, w*64+64):
// computes proj columns {g*C + j} for all 4 gates (16 WMMA tiles, 2 per wave),
// does the gate math + c update, then WGs 0..31 do the 16x512 projection
// (one N-tile each, per-wave split-K over 4096, LDS reduction).
// Ws/Wp are bf16 (16.8 MB / 4.2 MB per pass) -> L2-resident, so direct global
// B-fragment reads run at L2 bandwidth.
__global__ __launch_bounds__(256) void lstm_recurrent_kernel(
    const float* __restrict__ Xp, const __bf16* __restrict__ Wsb,
    const __bf16* __restrict__ Wpb, const int* __restrict__ seq_len,
    const float* __restrict__ resid, long resid_ld,
    float* __restrict__ out,
    float* __restrict__ h_state, __bf16* __restrict__ h_bf,
    float* __restrict__ c_state, __bf16* __restrict__ a_buf,
    unsigned* __restrict__ bar, int reverse, int has_res) {
  __shared__ float sm[16 * 256];                     // 16 tiles of 16x16 f32 (16 KB)
  __shared__ int slen[B_];
  const int wg = blockIdx.x;
  const int tid = threadIdx.x;
  const int lane = tid & 31, wave = tid >> 5;
  const bool hi = lane >= 16;
  const int nl = lane & 15;
  const int jbase = wg * 64;
  if (tid < B_) slen[tid] = seq_len[tid];
  __syncthreads();

  // this wave's two tile columns (fixed across timesteps)
  const int tau0 = wave * 2, tau1 = wave * 2 + 1;
  const int row0 = (tau0 >> 2) * C_ + jbase + (tau0 & 3) * 16 + nl;
  const int row1 = (tau1 >> 2) * C_ + jbase + (tau1 & 3) * 16 + nl;

  for (int t = 0; t < T_; ++t) {
    grid_sync(bar);                                  // h(t-1) visible everywhere

    // ---- hidden GEMM: acc = Xp[src_t] + h @ Ws.T for this WG's 256 gate cols
    v8f acc[2];
#pragma unroll
    for (int i = 0; i < 2; ++i) {
      const int col = (i == 0 ? row0 : row1);
#pragma unroll
      for (int v = 0; v < 8; ++v) {
        const int m = v + (hi ? 8 : 0);
        const int L = slen[m];
        const int src = reverse ? ((t < L) ? (L - 1 - t) : t) : t;
        acc[i][v] = Xp[((long)src * B_ + m) * G4C + col];
      }
    }
    for (int kb = 0; kb < H_; kb += 32) {
      if (kb + 32 < H_) {
        __builtin_prefetch(Wsb + (long)row0 * H_ + kb + 32, 0, 0);
        __builtin_prefetch(Wsb + (long)row1 * H_ + kb + 32, 0, 0);
      }
      // issue both B loads + A load before the WMMA pair
      v16bf b0 = load_b_bf16(Wsb, H_, row0, kb, hi);
      v16bf b1 = load_b_bf16(Wsb, H_, row1, kb, hi);
      v16bf a = load_a_bf16(h_bf, H_, nl, kb, hi);
      acc[0] = wmma_bf16(a, b0, acc[0]);
      acc[1] = wmma_bf16(a, b1, acc[1]);
    }
#pragma unroll
    for (int i = 0; i < 2; ++i) {
      float* dst = &sm[(wave * 2 + i) * 256];
#pragma unroll
      for (int v = 0; v < 8; ++v) dst[(v + (hi ? 8 : 0)) * 16 + nl] = acc[i][v];
    }
    __syncthreads();

    // ---- gate math + cell update for chunk [jbase, jbase+64), all 16 batch rows
#pragma unroll
    for (int it = 0; it < 4; ++it) {
      const int idx = tid + it * 256;                // 1024 = 16 b x 64 j
      const int b = idx >> 6, j = idx & 63;
      const int s = j >> 4, n = j & 15;
      const float xi = sm[((0 << 2) | s) * 256 + b * 16 + n];
      const float xf = sm[((1 << 2) | s) * 256 + b * 16 + n];
      const float xg = sm[((2 << 2) | s) * 256 + b * 16 + n];
      const float xo = sm[((3 << 2) | s) * 256 + b * 16 + n];
      const float ig = 1.f / (1.f + __expf(-xi));
      const float fg = 1.f / (1.f + __expf(-xf));
      const float gg = tanhf(xg);
      const float og = 1.f / (1.f + __expf(-xo));
      const long ci = (long)b * C_ + jbase + j;
      const float cold = c_state[ci];
      float cn = fminf(fmaxf(ig * gg + fg * cold, -3.f), 3.f);
      const bool valid = t < slen[b];
      c_state[ci] = valid ? cn : cold;
      a_buf[ci] = (__bf16)(og * tanhf(cn));          // h_new input, always from cn
    }
    grid_sync(bar);                                  // a_buf complete

    // ---- projection: h_new = clip(a @ Wp.T), WGs 0..31, one 16-col tile each
    if (wg < 32) {
      const int n0 = wg * 16;
      v8f pacc;
#pragma unroll
      for (int v = 0; v < 8; ++v) pacc[v] = 0.f;
      const int kb0 = wave * (C_ / 8);               // per-wave split-K (512 each)
      for (int kk = 0; kk < C_ / 8; kk += 32) {
        const int kb = kb0 + kk;
        v16bf b = load_b_bf16(Wpb, C_, n0 + nl, kb, hi);
        v16bf a = load_a_bf16(a_buf, C_, nl, kb, hi);
        pacc = wmma_bf16(a, b, pacc);
      }
#pragma unroll
      for (int v = 0; v < 8; ++v)
        sm[wave * 256 + (v + (hi ? 8 : 0)) * 16 + nl] = pacc[v];
      __syncthreads();
      {
        const int m = tid >> 4, n = tid & 15;        // 256 threads = 16x16 tile
        float sum = 0.f;
#pragma unroll
        for (int w = 0; w < 8; ++w) sum += sm[w * 256 + m * 16 + n];
        const float hn = fminf(fmaxf(sum, -3.f), 3.f);
        const int L = slen[m];
        const bool valid = t < L;
        const long hidx = (long)m * H_ + n0 + n;
        const float hm = valid ? hn : h_state[hidx];
        h_state[hidx] = hm;
        h_bf[hidx] = (__bf16)hm;
        // scatter output (reverse mapping is an involution; covers all p once)
        const int p = reverse ? (valid ? (L - 1 - t) : t) : t;
        float ov = valid ? hn : 0.f;
        if (has_res) ov += resid[((long)m * T_ + p) * resid_ld + n0 + n];
        out[((long)m * T_ + p) * OUT_LD + n0 + n] = ov;
      }
    }
  }
}

// ---------------------------------------------------------------- host driver
extern "C" void kernel_launch(void* const* d_in, const int* in_sizes, int n_in,
                              void* d_out, int out_size, void* d_ws, size_t ws_size,
                              hipStream_t stream) {
  (void)in_sizes; (void)n_in; (void)out_size; (void)ws_size;
  const float* x    = (const float*)d_in[0];
  const int*   slen = (const int*)d_in[1];
  const float* Wi   = (const float*)d_in[2];
  const float* Ws   = (const float*)d_in[3];
  const float* bia  = (const float*)d_in[4];
  const float* Wp   = (const float*)d_in[5];
  float* out = (float*)d_out;

  // workspace carve-up (~286 MB)
  char* p = (char*)d_ws;
  auto carve = [&](size_t bytes) -> char* {
    char* r = p; p += (bytes + 255) & ~(size_t)255; return r;
  };
  const size_t nWi = (size_t)L_ * 2 * G4C * D_;      // 33.5M
  const size_t nWs = (size_t)L_ * 2 * G4C * H_;      // 33.5M
  const size_t nWp = (size_t)L_ * 2 * H_ * C_;       // 8.4M
  __bf16* Wib = (__bf16*)carve(2 * nWi);
  __bf16* Wsb = (__bf16*)carve(2 * nWs);
  __bf16* Wpb = (__bf16*)carve(2 * nWp);
  float*  Xp  = (float*)carve(4ull * T_ * B_ * G4C); // 134 MB, reused per pass
  float*  hst = (float*)carve(4ull * B_ * H_);
  __bf16* hbf = (__bf16*)carve(2ull * B_ * H_);
  float*  cst = (float*)carve(4ull * B_ * C_);
  __bf16* abf = (__bf16*)carve(2ull * B_ * C_);
  unsigned* bar = (unsigned*)carve(256);

  // one-time (per launch) bf16 weight conversion — Ws bf16 is L2-resident per pass
  f32_to_bf16_kernel<<<4096, 256, 0, stream>>>(Wi, Wib, (long)nWi);
  f32_to_bf16_kernel<<<4096, 256, 0, stream>>>(Ws, Wsb, (long)nWs);
  f32_to_bf16_kernel<<<2048, 256, 0, stream>>>(Wp, Wpb, (long)nWp);

  for (int l = 0; l < L_; ++l) {
    for (int d = 0; d < 2; ++d) {
      const size_t wOff = (size_t)(2 * l + d) * G4C * H_;   // Wi/Ws (4C x 512)
      const size_t pOff = (size_t)(2 * l + d) * H_ * C_;    // Wp (512 x 4096)
      const size_t bOff = (size_t)(2 * l + d) * G4C;
      const float* in; long istr; const float* resid; int has_res;
      if (l == 0) { in = x; istr = D_; resid = nullptr; has_res = 0; }
      else {
        in = out + (size_t)(l - 1) * B_ * T_ * OUT_LD + (size_t)d * H_;
        istr = OUT_LD; resid = in; has_res = 1;
      }
      // 1) batched input projection for all timesteps (TDM-staged B panels)
      xproj_gemm_kernel<<<dim3(G4C / 128, (B_ * T_) / 128), 256, 0, stream>>>(
          in, istr, Wib + wOff, bia + bOff, Xp);
      // 2) reset recurrent state + barrier
      init_state_kernel<<<256, 256, 0, stream>>>(hst, hbf, cst, abf, bar);
      // 3) persistent recurrence
      lstm_recurrent_kernel<<<NWG, 256, 0, stream>>>(
          Xp, Wsb + wOff, Wpb + pOff, slen, resid, istr,
          out + (size_t)l * B_ * T_ * OUT_LD + (size_t)d * H_,
          hst, hbf, cst, abf, bar, /*reverse=*/d, has_res);
    }
  }
}
